// EntityExtraction_47854525612036
// MI455X (gfx1250) — matrix operand order repaired and checked by reading, not compiled, not run on hardware
//
#include <hip/hip_runtime.h>
#include <hip/hip_bf16.h>

#define B_   8
#define L_   512
#define DIM_ 1024
#define E_   64
#define K_   8
#define NE_  1000000

typedef _Float16 v8h  __attribute__((ext_vector_type(8)));
typedef _Float16 v16h __attribute__((ext_vector_type(16)));
typedef float    v8f  __attribute__((ext_vector_type(8)));

// ---------------------------------------------------------------- kernel 0: zero a float4 range
__global__ void k_zero(float* __restrict__ out, int n4) {
    int i = blockIdx.x * blockDim.x + threadIdx.x;
    if (i < n4) {
        float4 z = make_float4(0.f, 0.f, 0.f, 0.f);
        ((float4*)out)[i] = z;
    }
}

// ---------------------------------------------------------------- kernel 1: p[b,l] = q[b,l,:] . w
__global__ void k_rowdot(const float* __restrict__ q, const float* __restrict__ w,
                         float* __restrict__ p) {
    int bl = blockIdx.x;                 // b*L + l  (B*L blocks)
    int t  = threadIdx.x;                // 256 threads, 4 dims each
    const float* row = q + (size_t)bl * DIM_;
    float4 qv = *(const float4*)(row + t * 4);
    float4 wv = *(const float4*)(w   + t * 4);
    float s = qv.x * wv.x + qv.y * wv.y + qv.z * wv.z + qv.w * wv.w;
    __shared__ float red[256];
    red[t] = s; __syncthreads();
    for (int o = 128; o > 0; o >>= 1) { if (t < o) red[t] += red[t + o]; __syncthreads(); }
    if (t == 0) p[bl] = red[0];
}

// ---------------------------------------------------------------- kernel 2: span softmax -> s_sel
__global__ void k_span(const float* __restrict__ p, const int* __restrict__ span_i,
                       const int* __restrict__ span_j, float* __restrict__ s_sel) {
    int b = blockIdx.x;                  // B blocks, 256 threads
    int t = threadIdx.x;
    __shared__ float cs[L_];
    __shared__ float red[256];
    for (int l = t; l < L_; l += 256) cs[l] = p[b * L_ + l];
    __syncthreads();
    if (t == 0) {                         // inclusive cumsum (serial, 512 elems)
        float acc = 0.f;
        for (int l = 0; l < L_; ++l) { acc += cs[l]; cs[l] = acc; }
    }
    __syncthreads();
    // pass 1: max over valid spans i<j
    float mx = -__builtin_inff();
    for (int n = t; n < L_ * L_; n += 256) {
        int i = n >> 9, j = n & (L_ - 1);
        if (i < j) {
            float ci = (i == 0) ? 0.f : cs[i - 1];
            float sc = (cs[j] - ci) / (float)(j - i + 1);
            mx = fmaxf(mx, sc);
        }
    }
    red[t] = mx; __syncthreads();
    for (int o = 128; o > 0; o >>= 1) { if (t < o) red[t] = fmaxf(red[t], red[t + o]); __syncthreads(); }
    float gmax = red[0]; __syncthreads();
    // pass 2: sum of exp
    float se = 0.f;
    for (int n = t; n < L_ * L_; n += 256) {
        int i = n >> 9, j = n & (L_ - 1);
        if (i < j) {
            float ci = (i == 0) ? 0.f : cs[i - 1];
            float sc = (cs[j] - ci) / (float)(j - i + 1);
            se += __expf(sc - gmax);
        }
    }
    red[t] = se; __syncthreads();
    for (int o = 128; o > 0; o >>= 1) { if (t < o) red[t] += red[t + o]; __syncthreads(); }
    float gsum = red[0]; __syncthreads();
    if (t < E_) {
        int si = span_i[b * E_ + t], sj = span_j[b * E_ + t];
        float ci = (si == 0) ? 0.f : cs[si - 1];
        float sc = (cs[sj] - ci) / (float)(sj - si + 1);
        s_sel[b * E_ + t] = __expf(sc - gmax) / gsum;
    }
}

// ---------------------------------------------------------------- kernel 3: qspan (mean rows si..sj) -> f16
__global__ void k_qspan(const float* __restrict__ q, const int* __restrict__ span_i,
                        const int* __restrict__ span_j, _Float16* __restrict__ qspan_h) {
    int be = blockIdx.x;                 // b*E + e (512 blocks)
    int b  = be >> 6;                    // E_ = 64
    int t  = threadIdx.x;                // 256 threads, 4 dims each
    int si = span_i[be], sj = span_j[be];
    const float* base = q + (size_t)b * L_ * DIM_ + t * 4;
    float4 acc = make_float4(0.f, 0.f, 0.f, 0.f);
    for (int l = si; l <= sj; ++l) {
        float4 v = *(const float4*)(base + (size_t)l * DIM_);
        acc.x += v.x; acc.y += v.y; acc.z += v.z; acc.w += v.w;
    }
    float inv = 1.f / (float)(sj - si + 1);
    _Float16* o = qspan_h + (size_t)be * DIM_ + t * 4;
    o[0] = (_Float16)(acc.x * inv);
    o[1] = (_Float16)(acc.y * inv);
    o[2] = (_Float16)(acc.z * inv);
    o[3] = (_Float16)(acc.w * inv);
}

// ---------------------------------------------------------------- kernel 4: embedding-bag means -> f16
__global__ void k_zbags(const float* __restrict__ emb, const int* __restrict__ tokens,
                        _Float16* __restrict__ z_h) {
    int bek = blockIdx.x;                // b*E*K + .. (4096 blocks)
    int t   = threadIdx.x;               // 256 threads, 4 dims each
    int d   = t * 4;
    const int* tok = tokens + (size_t)bek * 16;   // G*T = 16 tokens per bag
    float4 acc = make_float4(0.f, 0.f, 0.f, 0.f);
    int tk_next = tok[0];
    for (int i = 0; i < 16; ++i) {
        int tk = tk_next;
        if (i < 15) {
            tk_next = tok[i + 1];
            __builtin_prefetch(emb + (size_t)tk_next * DIM_ + d, 0, 0);  // global_prefetch_b8
        }
        float4 v = *(const float4*)(emb + (size_t)tk * DIM_ + d);
        acc.x += v.x; acc.y += v.y; acc.z += v.z; acc.w += v.w;
    }
    const float inv = 1.f / 16.f;
    _Float16* o = z_h + (size_t)bek * DIM_ + d;
    o[0] = (_Float16)(acc.x * inv);
    o[1] = (_Float16)(acc.y * inv);
    o[2] = (_Float16)(acc.z * inv);
    o[3] = (_Float16)(acc.w * inv);
}

// ---------------------------------------------------------------- kernel 5: WMMA logits
// One wave (32 threads) per (b, entity-pair). A(16x32 f16) = 8 z-rows of e0 + 8 z-rows of e1,
// B(32x16 f16) = qspan(e0) in col 0, qspan(e1) in col 1; cols >= 2 read a zero page so all
// lanes load unconditionally (EXEC stays all-ones, no per-iter zero fills, loads clause).
// D layout (f32 16x16): lane 0 holds M=0..7,N=0 -> e0 logits; lane 17 holds M=8..15,N=1 -> e1.
__global__ void k_logits_wmma(const _Float16* __restrict__ z_h,
                              const _Float16* __restrict__ qs_h,
                              const _Float16* __restrict__ zpad,   // DIM_ halves of zeros
                              float* __restrict__ logits) {
    int wid  = blockIdx.x;               // B * (E/2) = 256 waves
    int b    = wid >> 5;                 // E/2 = 32
    int ep   = wid & 31;
    int e0   = ep * 2, e1 = e0 + 1;
    int lane = threadIdx.x;

    int row = lane & 15;                 // A-matrix row this lane supplies
    int e_r = (row < 8) ? e0 : e1;
    int k_r = row & 7;
    const _Float16* arow = z_h + (((size_t)b * E_ + e_r) * K_ + k_r) * DIM_;
    int khw = (lane < 16) ? 0 : 8;       // A half-lane K offset

    int col = lane & 15;                 // B-matrix column this lane supplies
    int k16 = (lane < 16) ? 0 : 8;       // B half-lane K offset
    const _Float16* bcol = (col < 2)
        ? (qs_h + ((size_t)b * E_ + ((col == 0) ? e0 : e1)) * DIM_)
        : zpad;                          // cols 2..15 read zeros (uniform, L0-broadcast)

    v8f c = {};
    for (int kc = 0; kc < DIM_; kc += 32) {
        v8h alo = *(const v8h*)(arow + kc + khw);
        v8h ahi = *(const v8h*)(arow + kc + 16 + khw);
        v16h a = __builtin_shufflevector(alo, ahi,
                                         0, 1, 2, 3, 4, 5, 6, 7, 8, 9, 10, 11, 12, 13, 14, 15);
        v8h blo = *(const v8h*)(bcol + kc + k16);
        v8h bhi = *(const v8h*)(bcol + kc + 16 + k16);
        v16h bm = __builtin_shufflevector(blo, bhi,
                                          0, 1, 2, 3, 4, 5, 6, 7, 8, 9, 10, 11, 12, 13, 14, 15);
        c = __builtin_amdgcn_wmma_f32_16x16x32_f16(false, a, false, bm, (short)0, c,
                                                   false, false);
    }
    if (lane == 0) {
        float* o = logits + ((size_t)b * E_ + e0) * K_;
        #pragma unroll
        for (int i = 0; i < 8; ++i) o[i] = c[i];
    }
    if (lane == 17) {
        float* o = logits + ((size_t)b * E_ + e1) * K_;
        #pragma unroll
        for (int i = 0; i < 8; ++i) o[i] = c[i];
    }
}

// ---------------------------------------------------------------- kernel 6: softmaxes + scatter
__global__ void k_final(const float* __restrict__ logits, const float* __restrict__ s_sel,
                        const int* __restrict__ idx, float* __restrict__ out) {
    int b = blockIdx.x;                  // B blocks, 256 threads
    int t = threadIdx.x;
    __shared__ float ebuf[E_ * K_];      // 512
    __shared__ float red[256];
    if (t < E_) {                        // per-entity softmax over K, scaled by s_sel
        const float* lg = logits + ((size_t)b * E_ + t) * K_;
        float mx = -__builtin_inff();
        #pragma unroll
        for (int k = 0; k < K_; ++k) mx = fmaxf(mx, lg[k]);
        float ex[K_]; float s = 0.f;
        #pragma unroll
        for (int k = 0; k < K_; ++k) { ex[k] = __expf(lg[k] - mx); s += ex[k]; }
        float w0 = s_sel[b * E_ + t] / s;
        #pragma unroll
        for (int k = 0; k < K_; ++k) ebuf[t * K_ + k] = ex[k] * w0;
    }
    __syncthreads();
    // global softmax over 512 values
    float m = fmaxf(ebuf[t], ebuf[t + 256]);
    red[t] = m; __syncthreads();
    for (int o = 128; o > 0; o >>= 1) { if (t < o) red[t] = fmaxf(red[t], red[t + o]); __syncthreads(); }
    float gmax = red[0]; __syncthreads();
    float s0 = __expf(ebuf[t] - gmax) + __expf(ebuf[t + 256] - gmax);
    red[t] = s0; __syncthreads();
    for (int o = 128; o > 0; o >>= 1) { if (t < o) red[t] += red[t + o]; __syncthreads(); }
    float gsum = red[0]; __syncthreads();
    for (int n = t; n < E_ * K_; n += 256) {
        float x = __expf(ebuf[n] - gmax) / gsum;
        out[(size_t)b * NE_ + idx[b * (E_ * K_) + n]] = x;
    }
}

// ---------------------------------------------------------------- launch
extern "C" void kernel_launch(void* const* d_in, const int* in_sizes, int n_in,
                              void* d_out, int out_size, void* d_ws, size_t ws_size,
                              hipStream_t stream) {
    const float* q      = (const float*)d_in[0];  // (B,L,DIM)
    const float* w      = (const float*)d_in[1];  // (DIM,)
    const float* emb    = (const float*)d_in[2];  // (N_PO,DIM)
    const int*   span_i = (const int*)d_in[3];    // (B,E)
    const int*   span_j = (const int*)d_in[4];    // (B,E)
    const int*   tokens = (const int*)d_in[5];    // (B,E,K,G,T)
    const int*   idx    = (const int*)d_in[6];    // (B,E*K)
    float*       out    = (float*)d_out;          // (B,N_E)

    char* ws = (char*)d_ws;
    // workspace layout (256B aligned)
    float*     p       = (float*)(ws + 0);                 // B*L f32       = 16 KB
    float*     s_sel   = (float*)(ws + 16384);             // B*E f32       =  2 KB
    float*     logits  = (float*)(ws + 18432);             // B*E*K f32     = 16 KB
    _Float16*  zpad    = (_Float16*)(ws + 34816);          // DIM f16 zeros =  2 KB
    _Float16*  qspan_h = (_Float16*)(ws + 36864);          // B*E*DIM f16   =  1 MB
    _Float16*  z_h     = (_Float16*)(ws + 1085440);        // B*E*K*DIM f16 =  8 MB

    int n4 = (B_ * NE_) / 4;
    k_zero<<<(n4 + 255) / 256, 256, 0, stream>>>(out, n4);
    k_zero<<<1, 128, 0, stream>>>((float*)zpad, (DIM_ * 2 / 4) / 4);  // 2 KB zero page
    k_rowdot<<<B_ * L_, 256, 0, stream>>>(q, w, p);
    k_span<<<B_, 256, 0, stream>>>(p, span_i, span_j, s_sel);
    k_qspan<<<B_ * E_, 256, 0, stream>>>(q, span_i, span_j, qspan_h);
    k_zbags<<<B_ * E_ * K_, 256, 0, stream>>>(emb, tokens, z_h);
    k_logits_wmma<<<B_ * (E_ / 2), 32, 0, stream>>>(z_h, qspan_h, zpad, logits);
    k_final<<<B_, 256, 0, stream>>>(logits, s_sel, idx, out);
}
